// EventTransformer_7095285973747
// MI455X (gfx1250) — matrix operand order repaired and checked by reading, not compiled, not run on hardware
//
#include <hip/hip_runtime.h>
#include <hip/hip_bf16.h>
#include <math.h>

typedef __attribute__((ext_vector_type(16))) _Float16 v16h;
typedef __attribute__((ext_vector_type(8)))  float    v8f;

static constexpr int B_ = 2, N_ = 8192, C_ = 64, M_ = 16, H_ = 180, W_ = 240;
static constexpr int HW_ = H_ * W_;
static constexpr long long BNC = (long long)B_ * N_ * C_;

// ---------------- utility ----------------
__global__ void zero_kernel(float* p, long long n) {
    for (long long i = blockIdx.x * (long long)blockDim.x + threadIdx.x; i < n;
         i += (long long)gridDim.x * blockDim.x)
        p[i] = 0.f;
}

// ---------------- lx_in = events @ W_mlp1 ----------------
__global__ void lx_in_kernel(const float* __restrict__ ev, const float* __restrict__ Wm,
                             float* __restrict__ lx_in) {
    long long idx = blockIdx.x * (long long)blockDim.x + threadIdx.x;
    if (idx >= BNC) return;
    int c = (int)(idx & 63);
    long long pt = idx >> 6;                 // b*N + n
    const float* e = ev + pt * 4;
    float s = e[0]*Wm[0*C_+c] + e[1]*Wm[1*C_+c] + e[2]*Wm[2*C_+c] + e[3]*Wm[3*C_+c];
    lx_in[idx] = s;
}

// ---------------- row sum over n: out[b,c] = sum_n X[b,n,c] ----------------
__global__ void rowsum_kernel(const float* __restrict__ X, float* __restrict__ out) {
    __shared__ float red[256];
    int bc = blockIdx.x;                     // b*C + c
    int b = bc / C_, c = bc % C_;
    float s = 0.f;
    for (int n = threadIdx.x; n < N_; n += 256)
        s += X[((long long)b * N_ + n) * C_ + c];
    red[threadIdx.x] = s; __syncthreads();
    for (int st = 128; st > 0; st >>= 1) {
        if (threadIdx.x < st) red[threadIdx.x] += red[threadIdx.x + st];
        __syncthreads();
    }
    if (threadIdx.x == 0) out[bc] = red[0];
}

// ---------------- tiny weight contractions ----------------
__global__ void prep_kernel(const float* lx_Wk, const float* lx_Wsa1, const float* W_pe,
                            const float* gx_Wpe, const float* gx_Wsa1,
                            float* wk1, float* wpe1, float* wpe1g) {
    int t = threadIdx.x;
    if (t < 64) {
        float a = 0.f, g = 0.f;
        for (int cn = 0; cn < 64; ++cn) {
            a += lx_Wk[t*64+cn] * lx_Wsa1[cn];
            g += gx_Wpe[t*64+cn] * gx_Wsa1[cn];
        }
        wk1[t] = a; wpe1g[t] = g;
    } else if (t < 68) {
        int j = t - 64; float a = 0.f;
        for (int cn = 0; cn < 64; ++cn) a += W_pe[j*64+cn] * lx_Wsa1[cn];
        wpe1[j] = a;
    }
}

// ke[b,n] = lx_in[b,n]·wk1 + events[b,n]·wpe1  (the only per-point logit term)
__global__ void ke_kernel(const float* __restrict__ ev, const float* __restrict__ lx_in,
                          const float* __restrict__ wk1, const float* __restrict__ wpe1,
                          float* __restrict__ ke) {
    int idx = blockIdx.x * blockDim.x + threadIdx.x;
    if (idx >= B_ * N_) return;
    float s = 0.f;
    const float* x = lx_in + (long long)idx * C_;
    for (int d = 0; d < 64; ++d) s += x[d] * wk1[d];
    const float* e = ev + (long long)idx * 4;
    for (int j = 0; j < 4; ++j) s += e[j] * wpe1[j];
    ke[idx] = s;
}

// S_loc[b,m,c] = windowed value-sum + windowed pe-sum (via total - head - tail)
__global__ void sloc_kernel(const float* __restrict__ ev, const float* __restrict__ lx_in,
                            const float* __restrict__ rs_in, const float* __restrict__ lx_Wv,
                            const float* __restrict__ W_pe, float* __restrict__ S_loc) {
    int idx = blockIdx.x * blockDim.x + threadIdx.x;
    if (idx >= B_ * M_ * C_) return;
    int c = idx & 63, m = (idx >> 6) & 15, b = idx >> 10;
    float sv = 0.f;
    for (int d = 0; d < 64; ++d) {
        float u = rs_in[b*C_+d];
        for (int j = 0; j < m - 8; ++j)        u -= lx_in[((long long)b*N_+j)*C_+d];
        for (int j = N_ - 8 + m; j < N_; ++j)  u -= lx_in[((long long)b*N_+j)*C_+d];
        sv += u * lx_Wv[d*C_+c];
    }
    float spe = 0.f;
    for (int j = 0; j < 4; ++j) {
        float he = 0.f;
        for (int jj = 0; jj < m - 8; ++jj)        he += ev[((long long)b*N_+jj)*4+j];
        for (int jj = N_ - 8 + m; jj < N_; ++jj)  he += ev[((long long)b*N_+jj)*4+j];
        spe += he * W_pe[j*C_+c];
    }
    S_loc[idx] = sv + spe;
}

// T[b,m,c'] = sum_c S[b,m,c] * Wsa2[c,c']   (shared by local and global branches)
__global__ void smallT_kernel(const float* __restrict__ S, const float* __restrict__ Wsa2,
                              float* __restrict__ T) {
    int idx = blockIdx.x * blockDim.x + threadIdx.x;
    if (idx >= B_ * M_ * C_) return;
    int cp = idx & 63; int bm = idx >> 6;
    float s = 0.f;
    for (int c = 0; c < 64; ++c) s += S[bm*C_+c] * Wsa2[c*C_+cp];
    T[idx] = s;
}

// local attention fuse: lx_out = lx_in + softmax_m(valid? -ke[j] : 0) @ T_loc
__global__ void local_fuse_kernel(const float* __restrict__ lx_in, const float* __restrict__ ke,
                                  const float* __restrict__ T_loc, float* __restrict__ lx_out) {
    __shared__ float lg[4][16];
    int t = threadIdx.x, g = t >> 6, lane = t & 63;
    int pt = blockIdx.x * 4 + g;
    int b = pt / N_, n = pt % N_;
    if (lane < 16) {
        int j = n + lane - 8;
        float v = 0.f;
        if (j >= 0 && j < N_) v = -ke[b*N_+j];
        lg[g][lane] = v;
    }
    __syncthreads();
    float mx = -1e30f;
    for (int m = 0; m < 16; ++m) mx = fmaxf(mx, lg[g][m]);
    float se = 0.f;
    for (int m = 0; m < 16; ++m) se += expf(lg[g][m] - mx);
    float o = lx_in[(long long)pt*C_ + lane];
    for (int m = 0; m < 16; ++m)
        o += (expf(lg[g][m] - mx) / se) * T_loc[(b*M_+m)*C_ + lane];
    lx_out[(long long)pt*C_ + lane] = o;
}

// ---------------- WMMA GEMM: kg = lx_out@gx_Wk, vg = lx_out@gx_Wv ----------------
__global__ void gemm_wmma_kernel(const float* __restrict__ lx_out,
                                 const float* __restrict__ gx_Wk,
                                 const float* __restrict__ gx_Wv,
                                 float* __restrict__ kg, float* __restrict__ vg) {
    const int id = (blockIdx.x * blockDim.x + threadIdx.x) >> 5;   // wave id
    const int lane = threadIdx.x & 31;
    const int tilesPer = (N_/16) * (C_/16);                        // 2048
    if (id >= B_ * 2 * tilesPer) return;
    int b = id / (2 * tilesPer);
    int r = id % (2 * tilesPer);
    int which = r / tilesPer;
    int t = r % tilesPer;
    int tm = t / (C_/16), tn = t % (C_/16);
    const float* X  = lx_out + (long long)b * N_ * C_;
    const float* Wm = which ? gx_Wv : gx_Wk;
    float*       Y  = (which ? vg : kg) + (long long)b * N_ * C_;
    const int hs = lane >> 4;          // half-lane select per ISA 16-bit layouts
    const int l  = lane & 15;
    const int row = tm*16 + l;
    const int col = tn*16 + l;
    v8f acc = {};
    #pragma unroll
    for (int kk = 0; kk < C_; kk += 32) {
        v16h a, bm;
        #pragma unroll
        for (int e = 0; e < 8; ++e) {      // A 16x32: lanes0-15 K{0..7,16..23}, lanes16-31 K{8..15,24..31}
            a[e]   = (_Float16)X[(long long)row*C_ + kk + hs*8 + e];
            a[e+8] = (_Float16)X[(long long)row*C_ + kk + 16 + hs*8 + e];
        }
        #pragma unroll
        for (int e = 0; e < 16; ++e)       // B 32x16: lanes0-15 K{0..7,16..23}, lanes16-31 K{8..15,24..31}
            bm[e] = (_Float16)Wm[(kk + (e >> 3)*16 + hs*8 + (e & 7))*C_ + col];
        acc = __builtin_amdgcn_wmma_f32_16x16x32_f16(false, a, false, bm, (short)0, acc,
                                                     false, false);
    }
    #pragma unroll
    for (int rr = 0; rr < 8; ++rr)          // D: VGPR r -> M = hs*8+r, N = l
        Y[(long long)(tm*16 + hs*8 + rr)*C_ + col] = acc[rr];
}

// ---------------- FPS (one block per batch x cloud) ----------------
__global__ void fps_kernel(const float* __restrict__ lx_out, const float* __restrict__ kg,
                           const float* __restrict__ vg, float* __restrict__ fL,
                           float* __restrict__ kfps, float* __restrict__ vfps) {
    constexpr int T = 512, PPT = N_ / T;
    __shared__ float part[T];
    __shared__ float cent[64];
    __shared__ float redv[T];
    __shared__ int   redi[T];
    __shared__ int   sel[16];
    int t = threadIdx.x;
    int b = blockIdx.x / 3, cloud = blockIdx.x % 3;
    const float* X = (cloud == 0 ? lx_out : cloud == 1 ? kg : vg) + (long long)b * N_ * C_;
    float* OUT = (cloud == 0 ? fL : cloud == 1 ? kfps : vfps) + (long long)b * M_ * C_;
    // barycenter
    {
        int c = t & 63, g = t >> 6;                       // 8 groups of 64 channels
        float s = 0.f;
        for (int i = 0; i < N_/8; ++i) s += X[((long long)(g*(N_/8)+i))*C_ + c];
        part[t] = s; __syncthreads();
        if (t < 64) {
            float a = 0.f;
            for (int g2 = 0; g2 < 8; ++g2) a += part[g2*64 + t];
            cent[t] = a / (float)N_;
        }
        __syncthreads();
    }
    float dist[PPT];
    float bv = -1e30f; int bi = 0x7fffffff;
    for (int i = 0; i < PPT; ++i) {
        int p = i * T + t;
        float d = 0.f;
        for (int c = 0; c < 64; ++c) { float df = X[(long long)p*C_+c] - cent[c]; d += df*df; }
        dist[i] = 1e10f;
        if (d > bv || (d == bv && p < bi)) { bv = d; bi = p; }
    }
    redv[t] = bv; redi[t] = bi; __syncthreads();
    for (int s = T/2; s > 0; s >>= 1) {
        if (t < s) {
            float v2 = redv[t+s]; int i2 = redi[t+s];
            if (v2 > redv[t] || (v2 == redv[t] && i2 < redi[t])) { redv[t] = v2; redi[t] = i2; }
        }
        __syncthreads();
    }
    if (t == 0) sel[0] = redi[0];
    __syncthreads();
    for (int s = 0; s < 15; ++s) {
        if (t < 64) cent[t] = X[(long long)sel[s]*C_ + t];
        __syncthreads();
        float lv = -1e30f; int li = 0x7fffffff;
        for (int i = 0; i < PPT; ++i) {
            int p = i * T + t;
            float d = 0.f;
            for (int c = 0; c < 64; ++c) { float df = X[(long long)p*C_+c] - cent[c]; d += df*df; }
            dist[i] = fminf(dist[i], d);
            if (dist[i] > lv || (dist[i] == lv && p < li)) { lv = dist[i]; li = p; }
        }
        redv[t] = lv; redi[t] = li; __syncthreads();
        for (int ss = T/2; ss > 0; ss >>= 1) {
            if (t < ss) {
                float v2 = redv[t+ss]; int i2 = redi[t+ss];
                if (v2 > redv[t] || (v2 == redv[t] && i2 < redi[t])) { redv[t] = v2; redi[t] = i2; }
            }
            __syncthreads();
        }
        if (t == 0) sel[s+1] = redi[0];
        __syncthreads();
    }
    if (t == 0) {                       // insertion sort (ascending) to match jnp.sort
        for (int i = 1; i < 16; ++i) {
            int key = sel[i], j = i - 1;
            while (j >= 0 && sel[j] > key) { sel[j+1] = sel[j]; --j; }
            sel[j+1] = key;
        }
    }
    __syncthreads();
    for (int idx = t; idx < M_ * C_; idx += T) {
        int m = idx >> 6, c = idx & 63;
        OUT[m*C_ + c] = X[(long long)sel[m]*C_ + c];
    }
}

// ---------------- global branch tiny kernels ----------------
__global__ void sg_kernel(const float* __restrict__ vfps, const float* __restrict__ fL,
                          const float* __restrict__ rs_out, const float* __restrict__ gx_Wpe,
                          float* __restrict__ S_g) {
    int idx = blockIdx.x * blockDim.x + threadIdx.x;
    if (idx >= B_ * M_ * C_) return;
    int c = idx & 63; int bm = idx >> 6; int b = bm >> 4;
    float s = (float)N_ * vfps[idx];
    for (int d = 0; d < 64; ++d)
        s += (rs_out[b*C_+d] - (float)N_ * fL[bm*C_+d]) * gx_Wpe[d*C_+c];
    S_g[idx] = s;
}

__global__ void mc_kernel(const float* __restrict__ kfps, const float* __restrict__ fL,
                          const float* __restrict__ gx_Wsa1, const float* __restrict__ wpe1g,
                          float* __restrict__ mc) {
    int idx = blockIdx.x * blockDim.x + threadIdx.x;
    if (idx >= B_ * M_) return;
    float s = 0.f;
    for (int cn = 0; cn < 64; ++cn)
        s += kfps[idx*C_+cn] * gx_Wsa1[cn] + fL[idx*C_+cn] * wpe1g[cn];
    mc[idx] = s;
}

// gconst[b,c] = softmax_m(-mc[b,:]) @ T_g[b,:,c]   (softmax shift-invariance!)
__global__ void gconst_kernel(const float* __restrict__ mc, const float* __restrict__ T_g,
                              float* __restrict__ gconst) {
    int idx = blockIdx.x * blockDim.x + threadIdx.x;
    if (idx >= B_ * C_) return;
    int b = idx >> 6, c = idx & 63;
    float mx = -1e30f;
    for (int m = 0; m < 16; ++m) mx = fmaxf(mx, -mc[b*M_+m]);
    float se = 0.f;
    for (int m = 0; m < 16; ++m) se += expf(-mc[b*M_+m] - mx);
    float o = 0.f;
    for (int m = 0; m < 16; ++m)
        o += (expf(-mc[b*M_+m] - mx) / se) * T_g[(b*M_+m)*C_ + c];
    gconst[idx] = o;
}

// ---------------- events_to_image ----------------
__global__ void scatter_kernel(const float* __restrict__ lx_out, const float* __restrict__ gconst,
                               const float* __restrict__ ev, const float* __restrict__ ln_w,
                               const float* __restrict__ ln_b, float* __restrict__ space) {
    __shared__ float xs[4][64];
    int t = threadIdx.x, g = t >> 6, c = t & 63;
    int pt = blockIdx.x * 4 + g;
    int b = pt / N_;
    float xv = lx_out[(long long)pt*C_ + c] + gconst[b*C_ + c];   // gx_out
    xs[g][c] = xv; __syncthreads();
    float mu = 0.f;
    for (int d = 0; d < 64; ++d) mu += xs[g][d];
    mu *= (1.f/64.f);
    float var = 0.f;
    for (int d = 0; d < 64; ++d) { float df = xs[g][d] - mu; var += df*df; }
    var *= (1.f/64.f);
    float xn = (xv - mu) * rsqrtf(var + 1e-5f) * ln_w[c] + ln_b[c];
    float h = 0.5f * xn * (1.f + erff(xn * 0.70710678118654752440f));   // exact GELU
    const float* e = ev + (long long)pt * 4;
    int xi = (int)floorf(e[0]);
    int yi = (int)floorf(e[1]);
    float p = e[3];
    int flat = yi * W_ + xi;
    atomicAdd(&space[((long long)b * HW_ + flat) * C_ + c], p * h);
}

// stats layout: [cnt0,sum0,cnt1,sum1,ssq0,ssq1,pad,pad]
__global__ void stats1_kernel(const float* __restrict__ space, float* __restrict__ stats) {
    const long long per = (long long)HW_ * C_;
    float c0 = 0.f, s0 = 0.f, c1 = 0.f, s1 = 0.f;
    for (long long i = blockIdx.x * (long long)blockDim.x + threadIdx.x; i < (long long)B_ * per;
         i += (long long)gridDim.x * blockDim.x) {
        float v = space[i];
        if (v != 0.f) {
            if (i < per) { c0 += 1.f; s0 += v; } else { c1 += 1.f; s1 += v; }
        }
    }
    atomicAdd(&stats[0], c0); atomicAdd(&stats[1], s0);
    atomicAdd(&stats[2], c1); atomicAdd(&stats[3], s1);
}

__global__ void stats2_kernel(const float* __restrict__ space, float* __restrict__ stats) {
    const long long per = (long long)HW_ * C_;
    float mean0 = stats[1] / fmaxf(stats[0], 1.f);
    float mean1 = stats[3] / fmaxf(stats[2], 1.f);
    float q0 = 0.f, q1 = 0.f;
    for (long long i = blockIdx.x * (long long)blockDim.x + threadIdx.x; i < (long long)B_ * per;
         i += (long long)gridDim.x * blockDim.x) {
        float v = space[i];
        if (v != 0.f) {
            if (i < per) { float d = v - mean0; q0 += d*d; }
            else         { float d = v - mean1; q1 += d*d; }
        }
    }
    atomicAdd(&stats[4], q0); atomicAdd(&stats[5], q1);
}

__global__ void finalize_kernel(const float* __restrict__ space, const float* __restrict__ stats,
                                float* __restrict__ out) {
    long long idx = blockIdx.x * (long long)blockDim.x + threadIdx.x;
    const long long tot = (long long)B_ * C_ * H_ * W_;
    if (idx >= tot) return;
    const long long chw = (long long)C_ * H_ * W_;
    int b = (int)(idx / chw);
    long long rem = idx % chw;
    int c = (int)(rem / HW_);
    int hw = (int)(rem % HW_);
    float v = space[((long long)b * HW_ + hw) * C_ + c];
    float cnt = stats[b*2], sum = stats[b*2+1], ssq = stats[4+b];
    float mean = sum / fmaxf(cnt, 1.f);
    float var  = ssq / fmaxf(cnt - 1.f, 1.f);
    float stdv = sqrtf(var);
    bool  mask = (v != 0.f);
    float centered = mask ? (v - mean) : v;
    float normed = (stdv > 0.f) ? (centered / stdv) : centered;
    out[idx] = (cnt > 0.f) ? normed : v;
}

// ---------------- launcher ----------------
extern "C" void kernel_launch(void* const* d_in, const int* in_sizes, int n_in,
                              void* d_out, int out_size, void* d_ws, size_t ws_size,
                              hipStream_t stream) {
    const float* events  = (const float*)d_in[0];
    const float* W_mlp1  = (const float*)d_in[1];
    const float* W_pe    = (const float*)d_in[2];
    const float* ln_w    = (const float*)d_in[3];
    const float* ln_b    = (const float*)d_in[4];
    const float* lx_Wk   = (const float*)d_in[6];
    const float* lx_Wv   = (const float*)d_in[7];
    const float* lx_Wsa1 = (const float*)d_in[8];
    const float* lx_Wsa2 = (const float*)d_in[9];
    const float* gx_Wk   = (const float*)d_in[11];
    const float* gx_Wv   = (const float*)d_in[12];
    const float* gx_Wpe  = (const float*)d_in[13];
    const float* gx_Wsa1 = (const float*)d_in[14];
    const float* gx_Wsa2 = (const float*)d_in[15];
    float* out = (float*)d_out;

    float* ws = (float*)d_ws;
    float* lx_in  = ws;                  ws += BNC;
    float* lx_out = ws;                  ws += BNC;
    float* kg     = ws;                  ws += BNC;
    float* vg     = ws;                  ws += BNC;
    float* space  = ws;                  ws += (long long)B_ * HW_ * C_;
    float* rs_in  = ws;                  ws += B_ * C_;
    float* rs_out = ws;                  ws += B_ * C_;
    float* ke     = ws;                  ws += B_ * N_;
    float* wk1    = ws;                  ws += 64;
    float* wpe1   = ws;                  ws += 8;
    float* wpe1g  = ws;                  ws += 64;
    float* S_loc  = ws;                  ws += B_ * M_ * C_;
    float* T_loc  = ws;                  ws += B_ * M_ * C_;
    float* fL     = ws;                  ws += B_ * M_ * C_;
    float* kfps   = ws;                  ws += B_ * M_ * C_;
    float* vfps   = ws;                  ws += B_ * M_ * C_;
    float* S_g    = ws;                  ws += B_ * M_ * C_;
    float* T_g    = ws;                  ws += B_ * M_ * C_;
    float* mc     = ws;                  ws += 32;
    float* gconst = ws;                  ws += B_ * C_;
    float* stats  = ws;                  ws += 8;

    // re-zero accumulation buffers every call
    zero_kernel<<<2048, 256, 0, stream>>>(space, (long long)B_ * HW_ * C_);
    zero_kernel<<<1, 32, 0, stream>>>(stats, 8);

    lx_in_kernel<<<(int)(BNC / 256), 256, 0, stream>>>(events, W_mlp1, lx_in);
    rowsum_kernel<<<B_ * C_, 256, 0, stream>>>(lx_in, rs_in);
    prep_kernel<<<1, 128, 0, stream>>>(lx_Wk, lx_Wsa1, W_pe, gx_Wpe, gx_Wsa1, wk1, wpe1, wpe1g);
    ke_kernel<<<(B_ * N_) / 256, 256, 0, stream>>>(events, lx_in, wk1, wpe1, ke);
    sloc_kernel<<<(B_ * M_ * C_) / 256, 256, 0, stream>>>(events, lx_in, rs_in, lx_Wv, W_pe, S_loc);
    smallT_kernel<<<(B_ * M_ * C_) / 256, 256, 0, stream>>>(S_loc, lx_Wsa2, T_loc);
    local_fuse_kernel<<<(B_ * N_) / 4, 256, 0, stream>>>(lx_in, ke, T_loc, lx_out);

    rowsum_kernel<<<B_ * C_, 256, 0, stream>>>(lx_out, rs_out);
    gemm_wmma_kernel<<<1024, 256, 0, stream>>>(lx_out, gx_Wk, gx_Wv, kg, vg);
    fps_kernel<<<6, 512, 0, stream>>>(lx_out, kg, vg, fL, kfps, vfps);

    sg_kernel<<<(B_ * M_ * C_) / 256, 256, 0, stream>>>(vfps, fL, rs_out, gx_Wpe, S_g);
    smallT_kernel<<<(B_ * M_ * C_) / 256, 256, 0, stream>>>(S_g, gx_Wsa2, T_g);
    mc_kernel<<<1, 64, 0, stream>>>(kfps, fL, gx_Wsa1, wpe1g, mc);
    gconst_kernel<<<1, 128, 0, stream>>>(mc, T_g, gconst);

    scatter_kernel<<<(B_ * N_) / 4, 256, 0, stream>>>(lx_out, gconst, events, ln_w, ln_b, space);
    stats1_kernel<<<256, 256, 0, stream>>>(space, stats);
    stats2_kernel<<<256, 256, 0, stream>>>(space, stats);
    const long long tot = (long long)B_ * C_ * H_ * W_;
    finalize_kernel<<<(int)((tot + 255) / 256), 256, 0, stream>>>(space, stats, out);
}
